// ResLSTMLayer_1975684956895
// MI455X (gfx1250) — matrix operand order repaired and checked by reading, not compiled, not run on hardware
//
#include <hip/hip_runtime.h>
#include <hip/hip_bf16.h>
#include <stdint.h>

// ResLSTMLayer for MI455X (gfx1250, wave32, WMMA).
// Phase 1: x_proj GEMM via v_wmma_f32_16x16x32_bf16 (fp32 accum), epilogue stored
//          in a [t][gatecol][batchrow] swizzle so phase 3 loads it with b128s.
// Phase 2: LN(x) residual.
// Phase 3: persistent recurrent kernel, 16 WGs, grid barriers per timestep.

#define T_STEPS 1024
#define BATCH   64
#define ISZ     512
#define HSZ     512
#define G4H     2048
#define EPSV    1e-5f

typedef __bf16 bf16;
typedef __attribute__((ext_vector_type(16))) __bf16 bf16x16;
typedef __attribute__((ext_vector_type(8)))  float  floatx8;

union BFrag { uint4 u[2]; bf16x16 v; };
union Pack8 { unsigned short s[8]; uint4 u; };

__device__ __forceinline__ unsigned short f2bf_bits(float f) {
    union { float f; unsigned u; } a; a.f = f;
    return (unsigned short)((a.u + 0x7FFFu + ((a.u >> 16) & 1u)) >> 16);   // RNE
}
__device__ __forceinline__ float bfbits2f(unsigned short s) {
    union { unsigned u; float f; } a; a.u = ((unsigned)s) << 16;
    return a.f;
}
__device__ __forceinline__ bf16 f2bf(float f) {
    union { unsigned short s; bf16 b; } o; o.s = f2bf_bits(f); return o.b;
}
__device__ __forceinline__ float sigf(float x) { return 1.0f / (1.0f + __expf(-x)); }

// ---------------------------------------------------------------- converts
__global__ void k_f32_to_bf16(const float* __restrict__ src, bf16* __restrict__ dst, int n) {
    int i = blockIdx.x * blockDim.x + threadIdx.x;
    int stride = gridDim.x * blockDim.x;
    for (; i < n; i += stride) dst[i] = f2bf(src[i]);
}

// h/c state init + zero sums + zero barrier (d_ws is poisoned by harness)
__global__ void k_init_state(const float* __restrict__ h0, const float* __restrict__ c0,
                             bf16* __restrict__ h, float* __restrict__ c,
                             float* __restrict__ sums, unsigned* __restrict__ bar) {
    int i = blockIdx.x * blockDim.x + threadIdx.x;
    if (i < BATCH * HSZ) { h[i] = f2bf(h0[i]); c[i] = c0[i]; }
    if (i < 256) sums[i] = 0.0f;     // 2 parities * (64 sum + 64 sumsq)
    if (i < 2)   bar[i]  = 0u;       // [0]=count [1]=generation
}

// ---------------------------------------------------------------- res = LN(x)
__global__ __launch_bounds__(256) void k_ln_res(const float* __restrict__ x,
                                                const float* __restrict__ g,
                                                const float* __restrict__ b,
                                                float* __restrict__ res) {
    const int row = blockIdx.x;                       // T*B rows
    const float* xr = x + (size_t)row * HSZ;
    float a0 = xr[threadIdx.x], a1 = xr[threadIdx.x + 256];
    float s = a0 + a1, q = a0 * a0 + a1 * a1;
    #pragma unroll
    for (int off = 16; off; off >>= 1) { s += __shfl_xor(s, off, 32); q += __shfl_xor(q, off, 32); }
    __shared__ float sh[16];
    const int widx = threadIdx.x >> 5, lane = threadIdx.x & 31;
    if (lane == 0) { sh[widx] = s; sh[8 + widx] = q; }
    __syncthreads();
    if (threadIdx.x == 0) {
        float ts = 0.f, tq = 0.f;
        #pragma unroll
        for (int w = 0; w < 8; ++w) { ts += sh[w]; tq += sh[8 + w]; }
        sh[0] = ts; sh[8] = tq;
    }
    __syncthreads();
    const float mean = sh[0] * (1.0f / HSZ);
    const float var  = sh[8] * (1.0f / HSZ) - mean * mean;
    const float rstd = rsqrtf(var + EPSV);
    float* rr = res + (size_t)row * HSZ;
    const int c0i = threadIdx.x, c1i = threadIdx.x + 256;
    rr[c0i] = (a0 - mean) * rstd * g[c0i] + b[c0i];
    rr[c1i] = (a1 - mean) * rstd * g[c1i] + b[c1i];
}

// ---------------------------------------------------------------- Phase 1 GEMM
// x_proj = x_bf16[65536,512] @ wih_bf16[2048,512]^T + bias
// WG = 8 waves; WG tile 128x128; wave tile 32x64 (2x4 WMMA tiles).
// Fragments loaded straight from global with b128s matching ISA 7.12.2 layouts.
// Epilogue: packed b128 stores into xp[t][gatecol][batchrow] swizzle.
__global__ __launch_bounds__(256) void k_xproj(const bf16* __restrict__ xb,
                                               const bf16* __restrict__ wih,
                                               const float* __restrict__ bias,
                                               bf16* __restrict__ xp) {
    const int lane = threadIdx.x & 31, widx = threadIdx.x >> 5;
    const int bm = blockIdx.x & 511, bn = blockIdx.x >> 9;
    const int rowbase = bm * 128 + (widx & 3) * 32;
    const int colbase = bn * 128 + (widx >> 2) * 64;
    const int half = lane >> 4, l16 = lane & 15;

    const floatx8 zero = {};
    floatx8 acc[2][4];
    #pragma unroll
    for (int mt = 0; mt < 2; ++mt)
        #pragma unroll
        for (int nt = 0; nt < 4; ++nt) acc[mt][nt] = zero;

    for (int kb = 0; kb < ISZ; kb += 32) {
        BFrag a[2], bf_[4];
        #pragma unroll
        for (int mt = 0; mt < 2; ++mt) {
            const bf16* p = xb + (size_t)(rowbase + mt * 16 + l16) * ISZ + kb + half * 8;
            a[mt].u[0] = *(const uint4*)p;
            a[mt].u[1] = *(const uint4*)(p + 16);
            __builtin_prefetch(p + 32, 0, 0);           // global_prefetch_b8 next K-chunk
        }
        #pragma unroll
        for (int nt = 0; nt < 4; ++nt) {
            const bf16* p = wih + (size_t)(colbase + nt * 16 + l16) * ISZ + kb + half * 8;
            bf_[nt].u[0] = *(const uint4*)p;
            bf_[nt].u[1] = *(const uint4*)(p + 16);
        }
        #pragma unroll
        for (int mt = 0; mt < 2; ++mt)
            #pragma unroll
            for (int nt = 0; nt < 4; ++nt)
                acc[mt][nt] = __builtin_amdgcn_wmma_f32_16x16x32_bf16(
                    false, a[mt].v, false, bf_[nt].v, (short)0, acc[mt][nt], false, false);
    }
    // Epilogue: per C-tile each lane owns (col fixed, 8 consecutive global rows).
    // Swizzled layout xp[t][gatecol][batchrow]: those 8 rows are contiguous -> b128.
    #pragma unroll
    for (int nt = 0; nt < 4; ++nt) {
        const int col = colbase + nt * 16 + l16;     // fixed per lane (C layout)
        const float bv = bias[col];
        #pragma unroll
        for (int mt = 0; mt < 2; ++mt) {
            const int rg0 = rowbase + mt * 16 + half * 8;    // first of 8 rows
            Pack8 pk;
            #pragma unroll
            for (int r = 0; r < 8; ++r) pk.s[r] = f2bf_bits(acc[mt][nt][r] + bv);
            const size_t idx = (size_t)(rg0 >> 6) * (G4H * BATCH)
                             + (size_t)col * BATCH + (rg0 & 63);
            *(uint4*)(xp + idx) = pk.u;
        }
    }
}

// ---------------------------------------------------------------- grid barrier
__device__ __forceinline__ void grid_barrier(unsigned* bar, unsigned nblk) {
    __syncthreads();
    if (threadIdx.x == 0) {
        __threadfence();
        unsigned g = __atomic_load_n(&bar[1], __ATOMIC_ACQUIRE);
        unsigned arrived = __atomic_fetch_add(&bar[0], 1u, __ATOMIC_ACQ_REL) + 1u;
        if (arrived == nblk) {
            __atomic_store_n(&bar[0], 0u, __ATOMIC_RELAXED);
            __atomic_fetch_add(&bar[1], 1u, __ATOMIC_ACQ_REL);
        } else {
            while (__atomic_load_n(&bar[1], __ATOMIC_ACQUIRE) == g)
                __builtin_amdgcn_s_sleep(1);
        }
        __threadfence();
    }
    __syncthreads();
}

// ---------------------------------------------------------------- Phase 3: recurrence
// Persistent: 16 WGs x 8 waves = 128 waves; wave owns one 16x16 hy tile
// (mt = batch-row tile 0..3, nt = hidden-col tile 0..31) and its 4 gate tiles.
// Per step: acc init from swizzled x_proj (4x b128), 16 K-chunks x 4 WMMA with
// all fragments issued before the WMMA group, LN(ry) via shfl-reduce + f32
// atomics, two grid barriers.
__global__ __launch_bounds__(256) void k_lstm_seq(
        const bf16* __restrict__ xp, const bf16* __restrict__ whh,
        const float* __restrict__ res, const float* __restrict__ lng,
        const float* __restrict__ lnb, bf16* h, float* c,
        float* sums, unsigned* bar, float* out) {
    const int lane = threadIdx.x & 31, widx = threadIdx.x >> 5;
    const int wg = blockIdx.x * 8 + widx;       // 0..127
    const int mt = wg & 3;                      // batch row tile
    const int nt = wg >> 2;                     // hidden col tile
    const int half = lane >> 4, l16 = lane & 15;
    const int col = nt * 16 + l16;              // fixed hidden column per lane
    const int row0 = mt * 16 + half * 8;        // first of this lane's 8 rows
    const float gv = lng[col], bvv = lnb[col];

    float* out_h  = out;                                    // [T,B,H]
    float* out_hf = out + (size_t)T_STEPS * BATCH * HSZ;    // [B,H]
    float* out_cf = out_hf + (size_t)BATCH * HSZ;           // [B,H]

    for (int t = 0; t < T_STEPS; ++t) {
        const int par = t & 1;
        floatx8 acc[4];
        // accumulator init = x-projection (bias folded) from swizzled layout:
        // one b128 load per gate (8 contiguous batch rows for this lane's col).
        #pragma unroll
        for (int g = 0; g < 4; ++g) {
            const size_t idx = (size_t)t * (G4H * BATCH)
                             + (size_t)(g * HSZ + col) * BATCH + row0;
            Pack8 pk; pk.u = *(const uint4*)(xp + idx);
            floatx8 z;
            #pragma unroll
            for (int r = 0; r < 8; ++r) z[r] = bfbits2f(pk.s[r]);
            acc[g] = z;
        }
        // gates += h @ whh^T   (whh + h resident in L2)
        #pragma unroll 2
        for (int kb = 0; kb < HSZ; kb += 32) {
            BFrag a, bf_[4];
            const bf16* pa = h + (size_t)(mt * 16 + l16) * HSZ + kb + half * 8;
            a.u[0] = *(const uint4*)pa;
            a.u[1] = *(const uint4*)(pa + 16);
            #pragma unroll
            for (int g = 0; g < 4; ++g) {
                const bf16* pb = whh + (size_t)(g * HSZ + col) * HSZ + kb + half * 8;
                bf_[g].u[0] = *(const uint4*)pb;
                bf_[g].u[1] = *(const uint4*)(pb + 16);
            }
            #pragma unroll
            for (int g = 0; g < 4; ++g)
                acc[g] = __builtin_amdgcn_wmma_f32_16x16x32_bf16(
                    false, a.v, false, bf_[g].v, (short)0, acc[g], false, false);
        }
        // gate nonlinearity + cell update (c is lane-owned: no hazard)
        float ry[8], og[8], cy[8];
        #pragma unroll
        for (int r = 0; r < 8; ++r) {
            const int row = row0 + r;
            const float ig = sigf(acc[0][r]);
            const float fg = sigf(acc[1][r]);
            const float gg = tanhf(acc[2][r]);
            og[r] = sigf(acc[3][r]);
            const float cx = c[(size_t)row * HSZ + col];
            const float cv = fg * cx + ig * gg;
            c[(size_t)row * HSZ + col] = cv;
            cy[r] = cv;
            ry[r] = tanhf(cv);
        }
        // per-batch-row sums for LN(ry): reduce 16 lanes of each half, then atomics
        #pragma unroll
        for (int r = 0; r < 8; ++r) {
            float s = ry[r], q = ry[r] * ry[r];
            #pragma unroll
            for (int off = 8; off; off >>= 1) {
                s += __shfl_xor(s, off, 32);
                q += __shfl_xor(q, off, 32);
            }
            if (l16 == 0) {
                atomicAdd(&sums[par * 128 + row0 + r], s);
                atomicAdd(&sums[par * 128 + 64 + row0 + r], q);
            }
        }
        grid_barrier(bar, gridDim.x);               // A: sums complete, h reads done
        // hy = o * (LN(ry) + res); publish next h
        #pragma unroll
        for (int r = 0; r < 8; ++r) {
            const int row  = row0 + r;
            const float mean = sums[par * 128 + row] * (1.0f / HSZ);
            const float var  = sums[par * 128 + 64 + row] * (1.0f / HSZ) - mean * mean;
            const float rstd = rsqrtf(var + EPSV);
            const float lnry = (ry[r] - mean) * rstd * gv + bvv;
            const float rv   = res[((size_t)(t * BATCH + row)) * HSZ + col];
            const float hy   = og[r] * (lnry + rv);
            out_h[((size_t)(t * BATCH + row)) * HSZ + col] = hy;
            h[(size_t)row * HSZ + col] = f2bf(hy);
            if (t == T_STEPS - 1) {
                out_hf[(size_t)row * HSZ + col] = hy;
                out_cf[(size_t)row * HSZ + col] = cy[r];
            }
        }
        // reset the other parity's sums for step t+1 (block 0 only)
        if (blockIdx.x == 0 && threadIdx.x < 128)
            sums[((t + 1) & 1) * 128 + threadIdx.x] = 0.0f;
        grid_barrier(bar, gridDim.x);               // B: h published, sums reset
    }
}

// ---------------------------------------------------------------- launch
extern "C" void kernel_launch(void* const* d_in, const int* in_sizes, int n_in,
                              void* d_out, int out_size, void* d_ws, size_t ws_size,
                              hipStream_t stream) {
    (void)in_sizes; (void)n_in; (void)out_size; (void)ws_size;
    const float* x     = (const float*)d_in[0];
    const float* h0    = (const float*)d_in[1];
    const float* c0    = (const float*)d_in[2];
    const float* wih   = (const float*)d_in[3];
    const float* whh   = (const float*)d_in[4];
    const float* bias  = (const float*)d_in[5];
    /* d_in[6] = weight_ir: unused in this branch of the reference */
    const float* lnryg = (const float*)d_in[7];
    const float* lnryb = (const float*)d_in[8];
    const float* lnresg= (const float*)d_in[9];
    const float* lnresb= (const float*)d_in[10];

    char* ws = (char*)d_ws;
    size_t off = 0;
    bf16*  xb   = (bf16*)(ws + off);  off += (size_t)T_STEPS * BATCH * ISZ * 2;  // 67 MB
    bf16*  wihb = (bf16*)(ws + off);  off += (size_t)G4H * ISZ * 2;              // 2 MB
    bf16*  whhb = (bf16*)(ws + off);  off += (size_t)G4H * HSZ * 2;              // 2 MB
    bf16*  xpb  = (bf16*)(ws + off);  off += (size_t)T_STEPS * BATCH * G4H * 2;  // 268 MB
    float* resf = (float*)(ws + off); off += (size_t)T_STEPS * BATCH * HSZ * 4;  // 134 MB
    bf16*  hbuf = (bf16*)(ws + off);  off += (size_t)BATCH * HSZ * 2;
    float* cbuf = (float*)(ws + off); off += (size_t)BATCH * HSZ * 4;
    float* sums = (float*)(ws + off); off += 256 * 4;
    unsigned* bar = (unsigned*)(ws + off); off += 64;

    k_f32_to_bf16<<<1024, 256, 0, stream>>>(x, xb, T_STEPS * BATCH * ISZ);
    k_f32_to_bf16<<<64,   256, 0, stream>>>(wih, wihb, G4H * ISZ);
    k_f32_to_bf16<<<64,   256, 0, stream>>>(whh, whhb, G4H * HSZ);
    k_init_state<<<(BATCH * HSZ + 255) / 256, 256, 0, stream>>>(h0, c0, hbuf, cbuf, sums, bar);
    k_ln_res<<<T_STEPS * BATCH, 256, 0, stream>>>(x, lnresg, lnresb, resf);
    k_xproj<<<8192, 256, 0, stream>>>(xb, wihb, bias, xpb);
    k_lstm_seq<<<16, 256, 0, stream>>>(xpb, whhb, resf, lnryg, lnryb, hbuf, cbuf, sums, bar,
                                       (float*)d_out);
}
